// STNAGNN_71588514890561
// MI455X (gfx1250) — matrix operand order repaired and checked by reading, not compiled, not run on hardware
//
#include <hip/hip_runtime.h>

#define NNODE 64512
#define NEDGE 1032192
#define ROI   84
#define WIN   12
#define HID   128
#define BATCH 64
#define SEQ   1008   /* ROI*WIN */

typedef _Float16 v16h __attribute__((ext_vector_type(16)));
typedef _Float16 v8h  __attribute__((ext_vector_type(8)));
typedef float    v8f  __attribute__((ext_vector_type(8)));

__device__ __forceinline__ v16h mk16(v8h lo, v8h hi) {
    v16h r;
#pragma unroll
    for (int i = 0; i < 8; ++i) { r[i] = lo[i]; r[i + 8] = hi[i]; }
    return r;
}

__device__ __forceinline__ v8f wmma_f16(v16h a, v16h b, v8f c) {
    return __builtin_amdgcn_wmma_f32_16x16x32_f16(false, a, false, b, (short)0, c,
                                                  false, false);
}

// ---------------------------------------------------------------- utilities
__global__ void f32_to_f16_kernel(const float* __restrict__ a, _Float16* __restrict__ o,
                                  unsigned n) {
    unsigned i = blockIdx.x * blockDim.x + threadIdx.x;
    if (i < n) o[i] = (_Float16)a[i];
}

// dst[row][c] = (c<K) ? src[row][c] : 0   (dst row stride = STRIDE floats)
__global__ void pad_copy_kernel(const float* __restrict__ src, float* __restrict__ dst,
                                int K, int stride, unsigned total) {
    unsigned i = blockIdx.x * blockDim.x + threadIdx.x;
    if (i >= total) return;
    unsigned row = i / stride, c = i % stride;
    dst[i] = ((int)c < K) ? src[(size_t)row * K + c] : 0.0f;
}

// W[k][n] (f32, KxNout) -> Wt[n][k] (f16, Nout x Kpad), zero pad k>=K
__global__ void conv_weight_kernel(const float* __restrict__ W, _Float16* __restrict__ Wt,
                                   int K, int Nout, int Kpad) {
    unsigned i = blockIdx.x * blockDim.x + threadIdx.x;
    if (i >= (unsigned)(Nout * Kpad)) return;
    int n = i / Kpad, k = i % Kpad;
    Wt[i] = (k < K) ? (_Float16)W[(size_t)k * Nout + n] : (_Float16)0.0f;
}

// ---------------------------------------------------------------- GIN scatter
// agg[dst] += feat[src] ; one thread per (edge, float4 chunk).
// feat row stride = sS4 float4s, agg row stride = dS4 float4s, F4 chunks added.
__global__ void scatter_add_kernel(const float* __restrict__ feat,
                                   float* __restrict__ agg,
                                   const long long* __restrict__ srcI,
                                   const long long* __restrict__ dstI,
                                   unsigned E, unsigned F4, unsigned sS4, unsigned dS4) {
    unsigned tid = blockIdx.x * blockDim.x + threadIdx.x;
    unsigned total = E * F4;
    if (tid >= total) return;
    unsigned e = tid / F4, c = tid % F4;
    long long s = srcI[e], d = dstI[e];
    const float4 v = ((const float4*)feat)[(size_t)s * sS4 + c];
    float* dp = agg + ((size_t)d * dS4 + c) * 4;
    atomicAdd(dp + 0, v.x);
    atomicAdd(dp + 1, v.y);
    atomicAdd(dp + 2, v.z);
    atomicAdd(dp + 3, v.w);
}

// ---------------------------------------------------------------- GIN GEMM
// C[M x 128] = A[M x KPAD](f32, zero-padded) @ W + bias ; Wt = f16 [128 x KPAD]
// one wave -> one 16-row M tile x all 8 N tiles.  grid = (M/16)/4 blocks of 128.
template <int KPAD>
__global__ __launch_bounds__(128) void gin_gemm_kernel(
    const float* __restrict__ A, const _Float16* __restrict__ Wt,
    const float* __restrict__ bias, float* __restrict__ C) {
    __shared__ _Float16 wt[128 * KPAD];
    __shared__ _Float16 at[4][16 * 32];

    const int lane = threadIdx.x & 31, wave = threadIdx.x >> 5;

    // CDNA5 async-LDS copy of the transposed weights (ASYNCcnt path):
    // each lane gathers one 16B chunk per issue straight into LDS.
    {
        unsigned ldsBase = (unsigned)(unsigned long long)(&wt[0]);
        const int nv8 = 128 * KPAD / 8;  // multiple of 128 -> full EXEC
#pragma unroll 4
        for (int i = threadIdx.x; i < nv8; i += 128) {
            unsigned loff = ldsBase + (unsigned)i * 16u;
            const _Float16* gp = Wt + (size_t)i * 8;
            asm volatile("global_load_async_to_lds_b128 %0, %1, off"
                         :: "v"(loff), "v"(gp)
                         : "memory");
        }
        asm volatile("s_wait_asynccnt 0x0" ::: "memory");
    }
    __syncthreads();

    const int mbase = (blockIdx.x * 4 + wave) * 16;
    const int r = lane & 15;
    const int halfsel = lane >> 4;
    const int col = lane & 15;
    const int cseg = halfsel * 16;
    const int kbA = halfsel ? 8 : 0;
    const int kbB = halfsel ? 8 : 0;

    v8f acc[8];
#pragma unroll
    for (int nt = 0; nt < 8; ++nt)
#pragma unroll
        for (int v = 0; v < 8; ++v) acc[nt][v] = 0.0f;

    const float* arow = A + (size_t)(mbase + r) * KPAD + cseg;

#pragma unroll
    for (int k0 = 0; k0 < KPAD; k0 += 32) {
        // stage 16x32 A tile f32 -> f16 : 4 unguarded float4 loads per lane
        const float4* ap4 = (const float4*)(arow + k0);
        float4 f0 = ap4[0], f1 = ap4[1], f2 = ap4[2], f3 = ap4[3];
        v8h lo, hi;
        lo[0] = (_Float16)f0.x; lo[1] = (_Float16)f0.y;
        lo[2] = (_Float16)f0.z; lo[3] = (_Float16)f0.w;
        lo[4] = (_Float16)f1.x; lo[5] = (_Float16)f1.y;
        lo[6] = (_Float16)f1.z; lo[7] = (_Float16)f1.w;
        hi[0] = (_Float16)f2.x; hi[1] = (_Float16)f2.y;
        hi[2] = (_Float16)f2.z; hi[3] = (_Float16)f2.w;
        hi[4] = (_Float16)f3.x; hi[5] = (_Float16)f3.y;
        hi[6] = (_Float16)f3.z; hi[7] = (_Float16)f3.w;
        v8h* adst = (v8h*)&at[wave][r * 32 + cseg];
        adst[0] = lo;
        adst[1] = hi;

        // A fragment (16x32 f16): halves = K {kbA..kbA+7, kbA+16..kbA+23}
        v16h afrag = mk16(*(const v8h*)&at[wave][r * 32 + kbA],
                          *(const v8h*)&at[wave][r * 32 + kbA + 16]);
#pragma unroll
        for (int nt = 0; nt < 8; ++nt) {
            const _Float16* wrow = &wt[(nt * 16 + col) * KPAD + k0 + kbB];
            v16h bfrag = mk16(*(const v8h*)wrow, *(const v8h*)(wrow + 16));
            acc[nt] = wmma_f16(afrag, bfrag, acc[nt]);
        }
    }
#pragma unroll
    for (int nt = 0; nt < 8; ++nt) {
        float bv = bias[nt * 16 + col];
#pragma unroll
        for (int v = 0; v < 8; ++v) {
            int row = mbase + v + 8 * halfsel;
            C[(size_t)row * HID + nt * 16 + col] = acc[nt][v] + bv;
        }
    }
}

// ---------------------------------------------------------------- q = t2 + pe
__global__ void make_q_kernel(const float* __restrict__ t2, const float* __restrict__ pe,
                              _Float16* __restrict__ qh, unsigned total) {
    unsigned i = blockIdx.x * blockDim.x + threadIdx.x;
    if (i >= total) return;
    if (i < (unsigned)NNODE * HID) {
        unsigned node = i / HID, d = i % HID;
        unsigned p = node % SEQ;
        qh[i] = (_Float16)(t2[i] + pe[p * HID + d]);
    } else {
        qh[i] = (_Float16)0.0f;  // pad rows
    }
}

// ---------------------------------------------------------------- flash attention
// one wave -> one 16-query tile; 63 key blocks of 16 (padded to K=32 in P@V).
__global__ __launch_bounds__(128) void attn_kernel(const _Float16* __restrict__ qh,
                                                   float* __restrict__ ctx) {
    __shared__ _Float16 pT[4][16 * 32];
    __shared__ _Float16 vT[4][128 * 32];

    const int lane = threadIdx.x & 31, wave = threadIdx.x >> 5;
    const int tile = blockIdx.x * 4 + wave;  // 0..4031
    const int b = tile / 63, qt = tile % 63;
    const int qbase = b * SEQ + qt * 16;
    const int kbatch = b * SEQ;
    const int r = lane & 15, halfsel = lane >> 4, col = lane & 15;
    const int kbA = halfsel ? 8 : 0;
    const int kbB = halfsel ? 8 : 0;

    _Float16* pt = pT[wave];
    _Float16* vt = vT[wave];

    // zero the padded halves once (P cols 16..31 and V-lds keys 16..31)
    v8h z;
#pragma unroll
    for (int t = 0; t < 8; ++t) z[t] = (_Float16)0.0f;
    *(v8h*)&pt[(lane >> 1) * 32 + 16 + (lane & 1) * 8] = z;
#pragma unroll
    for (int j = 0; j < 4; ++j) {
        int row = lane * 4 + j;
        *(v8h*)&vt[row * 32 + 16] = z;
        *(v8h*)&vt[row * 32 + 24] = z;
    }

    // Q fragments, resident for the whole tile (4 K-chunks of 32)
    v16h qfrag[4];
    const _Float16* qrow = qh + (size_t)(qbase + r) * HID;
#pragma unroll
    for (int c = 0; c < 4; ++c)
        qfrag[c] = mk16(*(const v8h*)(qrow + c * 32 + kbA),
                        *(const v8h*)(qrow + c * 32 + kbA + 16));

    v8f acc[8];
    float mrow[8], lrow[8];
#pragma unroll
    for (int nt = 0; nt < 8; ++nt)
#pragma unroll
        for (int v = 0; v < 8; ++v) acc[nt][v] = 0.0f;
#pragma unroll
    for (int v = 0; v < 8; ++v) { mrow[v] = -1e30f; lrow[v] = 0.0f; }

    const float scale = 0.0883883476483184f;  // 1/sqrt(128)

    for (int kblk = 0; kblk < 63; ++kblk) {
        const int keybase = kbatch + kblk * 16;

        // stage V transposed into LDS: vt[d][key]
        {
            int krow = lane >> 1;
            int dseg = (lane & 1) * 64;
            const _Float16* src = qh + (size_t)(keybase + krow) * HID + dseg;
#pragma unroll
            for (int j = 0; j < 8; ++j) {
                v8h val = *(const v8h*)(src + j * 8);
#pragma unroll
                for (int t = 0; t < 8; ++t) vt[(dseg + j * 8 + t) * 32 + krow] = val[t];
            }
        }

        // S = Q @ K^T  (B fragment: contiguous d from global)
        v8f sacc;
#pragma unroll
        for (int v = 0; v < 8; ++v) sacc[v] = 0.0f;
#pragma unroll
        for (int c = 0; c < 4; ++c) {
            const _Float16* kp = qh + (size_t)(keybase + col) * HID + c * 32 + kbB;
            v16h bfrag = mk16(*(const v8h*)kp, *(const v8h*)(kp + 16));
            sacc = wmma_f16(qfrag[c], bfrag, sacc);
        }

        // online softmax (rows live in registers: 8 rows per lane, C layout)
        float alpha[8], pval[8];
#pragma unroll
        for (int v = 0; v < 8; ++v) {
            float s = sacc[v] * scale;
            float rm = s;
            rm = fmaxf(rm, __shfl_xor(rm, 1, 32));
            rm = fmaxf(rm, __shfl_xor(rm, 2, 32));
            rm = fmaxf(rm, __shfl_xor(rm, 4, 32));
            rm = fmaxf(rm, __shfl_xor(rm, 8, 32));
            float nm = fmaxf(mrow[v], rm);
            float al = __expf(mrow[v] - nm);
            float p = __expf(s - nm);
            float rs = p;
            rs += __shfl_xor(rs, 1, 32);
            rs += __shfl_xor(rs, 2, 32);
            rs += __shfl_xor(rs, 4, 32);
            rs += __shfl_xor(rs, 8, 32);
            lrow[v] = lrow[v] * al + rs;
            mrow[v] = nm;
            alpha[v] = al;
            pval[v] = p;
        }
#pragma unroll
        for (int nt = 0; nt < 8; ++nt)
#pragma unroll
            for (int v = 0; v < 8; ++v) acc[nt][v] *= alpha[v];

        // write P tile (cols 16..31 stay zero), wave-synchronous
#pragma unroll
        for (int v = 0; v < 8; ++v)
            pt[(v + 8 * halfsel) * 32 + col] = (_Float16)pval[v];

        v16h pfrag = mk16(*(const v8h*)&pt[r * 32 + kbA],
                          *(const v8h*)&pt[r * 32 + kbA + 16]);

        // ctx += P @ V  (B fragments from transposed LDS staging)
#pragma unroll
        for (int nt = 0; nt < 8; ++nt) {
            const _Float16* vp = &vt[(nt * 16 + col) * 32 + kbB];
            v16h vfrag = mk16(*(const v8h*)vp, *(const v8h*)(vp + 16));
            acc[nt] = wmma_f16(pfrag, vfrag, acc[nt]);
        }
    }

#pragma unroll
    for (int nt = 0; nt < 8; ++nt)
#pragma unroll
        for (int v = 0; v < 8; ++v) {
            int row = qbase + v + 8 * halfsel;
            ctx[(size_t)row * HID + nt * 16 + col] = acc[nt][v] / lrow[v];
        }
}

// ---------------------------------------------------------------- pooling
__global__ void pool_kernel(const float* __restrict__ t1, const float* __restrict__ ctx,
                            float* __restrict__ feats) {
    unsigned i = blockIdx.x * blockDim.x + threadIdx.x;  // B*WIN*HID
    if (i >= (unsigned)BATCH * WIN * HID) return;
    int d = i % HID;
    int w = (i / HID) % WIN;
    int b = i / (HID * WIN);
    size_t base = ((size_t)b * SEQ + (size_t)w * ROI) * HID + d;
    float mx1 = -1e30f, s1 = 0.f, mx2 = -1e30f, s2 = 0.f;
    for (int rr = 0; rr < ROI; ++rr) {
        float a = t1[base + (size_t)rr * HID];
        float c = ctx[base + (size_t)rr * HID];
        mx1 = fmaxf(mx1, a);
        s1 += a;
        mx2 = fmaxf(mx2, c);
        s2 += c;
    }
    float* fo = feats + ((size_t)b * WIN + w) * (4 * HID);
    fo[d] = mx1;
    fo[HID + d] = s1 * (1.0f / ROI);
    fo[2 * HID + d] = mx2;
    fo[3 * HID + d] = s2 * (1.0f / ROI);
}

// ---------------------------------------------------------------- fc1 (WMMA) + silu + bn1
// out[64x512] ; A = feats f16 [64x6144], Wt = f16 [512x6144]
__global__ __launch_bounds__(128) void fc1_gemm_kernel(
    const _Float16* __restrict__ Ah, const _Float16* __restrict__ Wt,
    const float* __restrict__ bias, const float* __restrict__ g,
    const float* __restrict__ bb, const float* __restrict__ bm,
    const float* __restrict__ bv, float* __restrict__ outp) {
    const int lane = threadIdx.x & 31, wave = threadIdx.x >> 5;
    const int tid = blockIdx.x * 4 + wave;  // 0..127
    const int mtile = tid >> 5;             // 0..3
    const int ntile = tid & 31;             // 0..31
    const int r = lane & 15, halfsel = lane >> 4, col = lane & 15;
    const int kbA = halfsel ? 8 : 0;
    const int kbB = halfsel ? 8 : 0;

    v8f acc;
#pragma unroll
    for (int v = 0; v < 8; ++v) acc[v] = 0.0f;

    const _Float16* arow = Ah + (size_t)(mtile * 16 + r) * 6144;
    const _Float16* wrow = Wt + (size_t)(ntile * 16 + col) * 6144;
#pragma unroll 4
    for (int k0 = 0; k0 < 6144; k0 += 32) {
        v16h afrag = mk16(*(const v8h*)(arow + k0 + kbA),
                          *(const v8h*)(arow + k0 + kbA + 16));
        v16h bfrag = mk16(*(const v8h*)(wrow + k0 + kbB),
                          *(const v8h*)(wrow + k0 + kbB + 16));
        acc = wmma_f16(afrag, bfrag, acc);
    }
    int cg = ntile * 16 + col;
    float bvv = bias[cg];
    float inv = rsqrtf(bv[cg] + 1e-5f);
    float gg = g[cg], bbv = bb[cg], mm = bm[cg];
#pragma unroll
    for (int v = 0; v < 8; ++v) {
        int row = mtile * 16 + v + 8 * halfsel;
        float x = acc[v] + bvv;
        float s = x / (1.0f + __expf(-x));  // silu
        outp[(size_t)row * 512 + cg] = (s - mm) * inv * gg + bbv;
    }
}

// ---------------------------------------------------------------- fc2 + silu + bn2
__global__ void fc2_kernel(const float* __restrict__ h1, const float* __restrict__ w,
                           const float* __restrict__ bias, const float* __restrict__ g,
                           const float* __restrict__ bb, const float* __restrict__ bm,
                           const float* __restrict__ bv, float* __restrict__ h2) {
    unsigned i = blockIdx.x * blockDim.x + threadIdx.x;  // 64*32
    if (i >= BATCH * 32) return;
    int b = i >> 5, j = i & 31;
    float acc = bias[j];
    const float* hr = h1 + (size_t)b * 512;
    for (int k = 0; k < 512; ++k) acc += hr[k] * w[k * 32 + j];
    float s = acc / (1.0f + __expf(-acc));
    h2[i] = (s - bm[j]) * rsqrtf(bv[j] + 1e-5f) * g[j] + bb[j];
}

// ---------------------------------------------------------------- fc3 + softmax
__global__ void fc3_kernel(const float* __restrict__ h2, const float* __restrict__ w,
                           const float* __restrict__ bias, float* __restrict__ outp) {
    unsigned b = blockIdx.x * blockDim.x + threadIdx.x;
    if (b >= BATCH) return;
    float z0 = bias[0], z1 = bias[1];
    for (int j = 0; j < 32; ++j) {
        float h = h2[b * 32 + j];
        z0 += h * w[j * 2 + 0];
        z1 += h * w[j * 2 + 1];
    }
    float mx = fmaxf(z0, z1);
    float e0 = __expf(z0 - mx), e1 = __expf(z1 - mx);
    float s = e0 + e1;
    outp[b * 2 + 0] = e0 / s;
    outp[b * 2 + 1] = e1 / s;
}

// ================================================================ launch
extern "C" void kernel_launch(void* const* d_in, const int* in_sizes, int n_in,
                              void* d_out, int out_size, void* d_ws, size_t ws_size,
                              hipStream_t stream) {
    (void)in_sizes; (void)n_in; (void)out_size; (void)ws_size;

    const float* x      = (const float*)d_in[0];
    const long long* ei = (const long long*)d_in[1];
    const float* pe     = (const float*)d_in[2];
    const float* W1     = (const float*)d_in[3];
    const float* b1     = (const float*)d_in[4];
    const float* W2     = (const float*)d_in[5];
    const float* b2     = (const float*)d_in[6];
    const float* fc1_w  = (const float*)d_in[7];
    const float* fc1_b  = (const float*)d_in[8];
    const float* fc2_w  = (const float*)d_in[9];
    const float* fc2_b  = (const float*)d_in[10];
    const float* fc3_w  = (const float*)d_in[11];
    const float* fc3_b  = (const float*)d_in[12];
    const float* bn1_g  = (const float*)d_in[13];
    const float* bn1_b  = (const float*)d_in[14];
    const float* bn1_m  = (const float*)d_in[15];
    const float* bn1_v  = (const float*)d_in[16];
    const float* bn2_g  = (const float*)d_in[17];
    const float* bn2_b  = (const float*)d_in[18];
    const float* bn2_m  = (const float*)d_in[19];
    const float* bn2_v  = (const float*)d_in[20];
    float* out = (float*)d_out;

    const long long* srcI = ei;
    const long long* dstI = ei + NEDGE;

    char* ws = (char*)d_ws;
    size_t off = 0;
    auto alloc = [&](size_t bytes) -> char* {
        char* p = ws + off;
        off = (off + bytes + 255) & ~(size_t)255;
        return p;
    };
    float*    agg1   = (float*)alloc((size_t)NNODE * 96 * 4);   // padded 84 -> 96
    float*    t1     = (float*)alloc((size_t)NNODE * HID * 4);
    float*    agg2   = (float*)alloc((size_t)NNODE * HID * 4);
    float*    t2     = (float*)alloc((size_t)NNODE * HID * 4);
    float*    ctxb   = (float*)alloc((size_t)NNODE * HID * 4);
    _Float16* qh     = (_Float16*)alloc((size_t)(NNODE + 32) * HID * 2);
    _Float16* w1t    = (_Float16*)alloc((size_t)128 * 96 * 2);
    _Float16* w2t    = (_Float16*)alloc((size_t)128 * 128 * 2);
    _Float16* fc1t   = (_Float16*)alloc((size_t)512 * 6144 * 2);
    float*    featsF = (float*)alloc((size_t)BATCH * 6144 * 4);
    _Float16* featsH = (_Float16*)alloc((size_t)BATCH * 6144 * 2);
    float*    h1     = (float*)alloc((size_t)BATCH * 512 * 4);
    float*    h2     = (float*)alloc((size_t)BATCH * 32 * 4);

    const int TPB = 256;
    auto g1 = [&](unsigned n) { return dim3((n + TPB - 1) / TPB); };

    // weight prep (f32 -> f16 transposed, K zero-padded)
    conv_weight_kernel<<<g1(128 * 96), TPB, 0, stream>>>(W1, w1t, ROI, HID, 96);
    conv_weight_kernel<<<g1(128 * 128), TPB, 0, stream>>>(W2, w2t, HID, HID, 128);
    conv_weight_kernel<<<g1(512 * 6144), TPB, 0, stream>>>(fc1_w, fc1t, 6144, 512, 6144);

    // GIN layer 1: agg1 = x (padded to stride 96) + scatter(x) ; t1 = agg1 @ W1 + b1
    pad_copy_kernel<<<g1((unsigned)NNODE * 96), TPB, 0, stream>>>(
        x, agg1, ROI, 96, (unsigned)NNODE * 96);
    scatter_add_kernel<<<g1((unsigned)NEDGE * 21), TPB, 0, stream>>>(
        x, agg1, srcI, dstI, NEDGE, 21, 21, 24);
    gin_gemm_kernel<96><<<dim3(NNODE / 64), dim3(128), 0, stream>>>(agg1, w1t, b1, t1);

    // GIN layer 2
    pad_copy_kernel<<<g1((unsigned)NNODE * HID), TPB, 0, stream>>>(
        t1, agg2, HID, HID, (unsigned)NNODE * HID);
    scatter_add_kernel<<<g1((unsigned)NEDGE * 32), TPB, 0, stream>>>(
        t1, agg2, srcI, dstI, NEDGE, 32, 32, 32);
    gin_gemm_kernel<128><<<dim3(NNODE / 64), dim3(128), 0, stream>>>(agg2, w2t, b2, t2);

    // q = t2 + pe (f16), padded tail zeroed
    make_q_kernel<<<g1((unsigned)(NNODE + 32) * HID), TPB, 0, stream>>>(
        t2, pe, qh, (unsigned)(NNODE + 32) * HID);

    // flash self-attention: 4032 query tiles, 4 waves/block
    attn_kernel<<<dim3(NNODE / 64), dim3(128), 0, stream>>>(qh, ctxb);

    // pooling + head
    pool_kernel<<<g1(BATCH * WIN * HID), TPB, 0, stream>>>(t1, ctxb, featsF);
    f32_to_f16_kernel<<<g1(BATCH * 6144), TPB, 0, stream>>>(featsF, featsH, BATCH * 6144);
    fc1_gemm_kernel<<<dim3(32), dim3(128), 0, stream>>>(featsH, fc1t, fc1_b, bn1_g, bn1_b,
                                                        bn1_m, bn1_v, h1);
    fc2_kernel<<<g1(BATCH * 32), TPB, 0, stream>>>(h1, fc2_w, fc2_b, bn2_g, bn2_b, bn2_m,
                                                   bn2_v, h2);
    fc3_kernel<<<dim3(1), dim3(64), 0, stream>>>(h2, fc3_w, fc3_b, out);
}
